// CrossAttentionBlock_20993800143080
// MI455X (gfx1250) — matrix operand order repaired and checked by reading, not compiled, not run on hardware
//
#include <hip/hip_runtime.h>

typedef __attribute__((ext_vector_type(16))) __bf16 bf16x16;
typedef __attribute__((ext_vector_type(8)))  __bf16 bf16x8;
typedef __attribute__((ext_vector_type(4)))  __bf16 bf16x4;
typedef __attribute__((ext_vector_type(8)))  float  f32x8;
typedef __attribute__((ext_vector_type(4)))  unsigned int v4u;
typedef __attribute__((ext_vector_type(8)))  int v8i_t;
typedef __attribute__((ext_vector_type(4)))  int v4i_t;

#define DIM_   192
#define DH_    96
#define NHEAD_ 6
#define HID_   384
#define S_C    16
#define H_C    32
#define W_C    32
#define S_F    32
#define H_F    64
#define W_F    64
#define NQ_    512
#define NK_    64
#define SCALE_ 0.17677669529663687f

// bf16 [N][K] weight sizes (elements)
#define SZ_KV  (384 * 192)
#define SZ_Q   (192 * 96)
#define SZ_P   (96 * 192)
#define SZ_1   (384 * 96)
#define SZ_2   (96 * 384)
#define SZ_DW  (8 * 96 * 192)
#define SZ_WT_TOTAL (SZ_KV + SZ_Q + SZ_P + SZ_1 + SZ_2 + SZ_DW)

#if defined(__has_builtin)
#if __has_builtin(__builtin_amdgcn_tensor_load_to_lds) && __has_builtin(__builtin_amdgcn_s_wait_tensorcnt)
#define HAVE_TDM 1
#endif
#endif

static __device__ __forceinline__ f32x8 zero8() {
  f32x8 z;
#pragma unroll
  for (int e = 0; e < 8; ++e) z[e] = 0.0f;
  return z;
}

static __device__ __forceinline__ f32x8 wmma_bf16(bf16x16 a, bf16x16 b, f32x8 c) {
  return __builtin_amdgcn_wmma_f32_16x16x32_bf16(false, a, false, b, (short)0, c, false, false);
}

// A operand (16x32 bf16) from row-major [M][K] bf16 buffer.
static __device__ __forceinline__ bf16x16 load_a(const __bf16* base, int ld, int rowBase, int kBase, int lid) {
  int m = lid & 15, half = lid >> 4;
  const __bf16* p = base + (size_t)(rowBase + m) * ld + kBase + half * 8;
  union { bf16x16 v; bf16x8 h[2]; } u;
  u.h[0] = *(const bf16x8*)(p);
  u.h[1] = *(const bf16x8*)(p + 16);
  return u.v;
}

// B operand (32x16 bf16) from an N-major buffer laid out [N][K] (ldk = K stride).
static __device__ __forceinline__ bf16x16 load_b_nk(const __bf16* base, int ldk, int colBase, int kBase, int lid) {
  int m = lid & 15, half = lid >> 4;
  const __bf16* p = base + (size_t)(colBase + m) * ldk + kBase + half * 8;
  union { bf16x16 v; bf16x8 h[2]; } u;
  u.h[0] = *(const bf16x8*)(p);
  u.h[1] = *(const bf16x8*)(p + 16);
  return u.v;
}

#ifdef HAVE_TDM
// Issue a 1-D TDM copy: nElems f32 from global -> LDS. Must be wave-uniform.
static __device__ __forceinline__ void tdm_copy_f32(unsigned int ldsOff, const float* gsrc, int nElems) {
  unsigned long long ga = (unsigned long long)(uintptr_t)gsrc;
  v4u g0;
  g0[0] = 1u;                                            // count=1, user descriptor
  g0[1] = ldsOff;                                        // lds_addr
  g0[2] = (unsigned int)ga;                              // global_addr[31:0]
  g0[3] = (unsigned int)((ga >> 32) & 0x1FFFFFFu) | (2u << 30);  // addr[56:32] | type=2
  v8i_t g1;
  g1[0] = (2 << 16);                                     // data_size = 4B
  g1[1] = (int)((unsigned)nElems << 16);                 // tensor_dim0[15:0]
  g1[2] = (int)(((unsigned)nElems >> 16) & 0xFFFFu) | (1 << 16);  // tensor_dim0 hi | tensor_dim1=1
  g1[3] = (int)((unsigned)nElems << 16);                 // tensor_dim1 hi=0 | tile_dim0
  g1[4] = 1;                                             // tile_dim1=1, tile_dim2=0
  g1[5] = nElems;                                        // tensor_dim0_stride lo
  g1[6] = 0;
  g1[7] = 0;
  v4i_t gz;
  gz[0] = 0; gz[1] = 0; gz[2] = 0; gz[3] = 0;
#if __clang_major__ >= 23
  v8i_t gz8;
#pragma unroll
  for (int e = 0; e < 8; ++e) gz8[e] = 0;
  __builtin_amdgcn_tensor_load_to_lds(g0, g1, gz, gz, gz8, 0);
#else
  __builtin_amdgcn_tensor_load_to_lds(g0, g1, gz, gz, 0);
#endif
}
#endif

// ---------------------------------------------------------------------------
// Kernel 0: weight prep (f32 [K][N] -> bf16 [N][K]) + bias table expansion.
// ---------------------------------------------------------------------------
__global__ void prep_kernel(const float* __restrict__ Wkv, const float* __restrict__ Wq,
                            const float* __restrict__ Wp,  const float* __restrict__ W1,
                            const float* __restrict__ W2,  const float* __restrict__ dw,
                            const float* __restrict__ table,
                            __bf16* __restrict__ wtBase, float* __restrict__ biasF) {
  int id = blockIdx.x * blockDim.x + threadIdx.x;
  if (id < NHEAD_ * NQ_ * NK_) {
    int kk = id & 63;
    int q  = (id >> 6) & 511;
    int h  = id >> 15;
    int qs = (q >> 6) >> 1;
    int qh = ((q >> 3) & 7) >> 1;
    int qw = (q & 7) >> 1;
    int ks = kk >> 4, kh = (kk >> 2) & 3, kw = kk & 3;
    int rel = ((qs - ks + 3) * 7 + (qh - kh + 3)) * 7 + (qw - kw + 3);
    biasF[id] = table[rel * NHEAD_ + h];
  }
  if (id >= SZ_WT_TOTAL) return;
  int i = id;
  if (i < SZ_KV) { int n = i / 192, k = i - n * 192; wtBase[id] = (__bf16)Wkv[k * 384 + n]; return; }
  i -= SZ_KV;
  if (i < SZ_Q)  { int n = i / 96, k = i - n * 96;   wtBase[id] = (__bf16)Wq[k * 192 + n]; return; }
  i -= SZ_Q;
  if (i < SZ_P)  { int n = i / 192, k = i - n * 192; wtBase[id] = (__bf16)Wp[k * 96 + n]; return; }
  i -= SZ_P;
  if (i < SZ_1)  { int n = i / 96, k = i - n * 96;   wtBase[id] = (__bf16)W1[k * 384 + n]; return; }
  i -= SZ_1;
  if (i < SZ_2)  { int n = i / 384, k = i - n * 384; wtBase[id] = (__bf16)W2[k * 96 + n]; return; }
  i -= SZ_2;
  { int p = i / (96 * 192); int r = i - p * 96 * 192; int n = r / 192, k = r - n * 192;
    wtBase[id] = (__bf16)dw[((size_t)k * 96 + n) * 8 + p]; }
}

// ---------------------------------------------------------------------------
// Kernel 1: windowed cross attention. 1 block = 1 window. 256 threads = 8 waves.
// ---------------------------------------------------------------------------
__launch_bounds__(256, 1)
__global__ void attn_kernel(const float* __restrict__ x, const float* __restrict__ xd,
                            const float* __restrict__ g1, const float* __restrict__ be1,
                            const __bf16* __restrict__ WkvT, const float* __restrict__ bkv,
                            const __bf16* __restrict__ WqT,  const float* __restrict__ bq,
                            const __bf16* __restrict__ WpT,  const float* __restrict__ bp,
                            const float* __restrict__ biasF, float* __restrict__ attnO) {
  __shared__ alignas(16) char smem[151552];
  __bf16* kS  = (__bf16*)(smem);            // [64][192] k, row-major [kk][d]
  __bf16* vT  = (__bf16*)(smem + 24576);    // [192][64] v transposed [d][kk]
  __bf16* kvA = (__bf16*)(smem + 49152);    // [64][192] LN'ed kv window (phase 2 only)
  __bf16* qIn = (__bf16*)(smem + 49152);    // [64][96]  staged raw q tile (aliases kvA)
  __bf16* qS  = (__bf16*)(smem + 61440);    // [64][192] q after projection*scale
  float*  aS  = (float*)(smem + 86016);     // [64][64]  attention scores
  __bf16* pS  = (__bf16*)(smem + 102400);   // [64][64]  softmax probs
  __bf16* oS  = (__bf16*)(smem + 110592);   // [64][192] per-tile attention output
  float*  bS  = (float*)(smem + 135168);    // [64][64]  bias slab (TDM dest)

  const int tid = threadIdx.x, wave = tid >> 5, lid = tid & 31;
  const int wid = blockIdx.x;
  const int bat = wid >> 8;
  const int rem = wid & 255;
  const int isb = rem >> 6, ihb = (rem >> 3) & 7, iwb = rem & 7;
  const int s0 = isb * 4, h0 = ihb * 4, w0 = iwb * 4;
  const int sd0 = isb * 8, hd0 = ihb * 8, wd0 = iwb * 8;
  const int m = lid & 15, half = lid >> 4;

  // ---- Phase 1: LN of the coarse window (64 rows x 192) -> kvA (bf16) ----
  for (int n = wave * 8; n < wave * 8 + 8; ++n) {
    int is = n >> 4, ih = (n >> 2) & 3, iw = n & 3;
    const float* row = x + ((size_t)bat * S_C * H_C * W_C +
                            ((size_t)(s0 + is) * H_C + (h0 + ih)) * W_C + (w0 + iw)) * DIM_;
    float sm = 0.f, ss = 0.f, vb[6];
#pragma unroll
    for (int j = 0; j < 6; ++j) { float v = row[lid + 32 * j]; vb[j] = v; sm += v; ss += v * v; }
#pragma unroll
    for (int off = 16; off; off >>= 1) { sm += __shfl_xor(sm, off); ss += __shfl_xor(ss, off); }
    float mu = sm * (1.f / 192.f);
    float rstd = rsqrtf(ss * (1.f / 192.f) - mu * mu + 1e-5f);
#pragma unroll
    for (int j = 0; j < 6; ++j) {
      int c = lid + 32 * j;
      kvA[n * DIM_ + c] = (__bf16)((vb[j] - mu) * rstd * g1[c] + be1[c]);
    }
  }
  __syncthreads();

  // ---- Phase 2a: k (transposed orientation): C'[d][kk] = WkvT(rows 0..191) x kvA^T ----
  for (int t = wave; t < 48; t += 8) {
    int mi = t >> 2, ni = t & 3;        // mi: d tile (12), ni: kk tile (4)
    f32x8 acc = zero8();
#pragma unroll
    for (int kb = 0; kb < 6; ++kb) {
      bf16x16 a = load_a(WkvT, DIM_, mi * 16, kb * 32, lid);
      bf16x16 b = load_b_nk(kvA, DIM_, ni * 16, kb * 32, lid);
      acc = wmma_bf16(a, b, acc);
    }
    int rowB = mi * 16 + half * 8;      // d base (multiple of 8)
    int col = ni * 16 + m;              // kk
    const float4 bv0 = *(const float4*)(bkv + rowB);
    const float4 bv1 = *(const float4*)(bkv + rowB + 4);
    bf16x8 pk;
    pk[0] = (__bf16)(acc[0] + bv0.x); pk[1] = (__bf16)(acc[1] + bv0.y);
    pk[2] = (__bf16)(acc[2] + bv0.z); pk[3] = (__bf16)(acc[3] + bv0.w);
    pk[4] = (__bf16)(acc[4] + bv1.x); pk[5] = (__bf16)(acc[5] + bv1.y);
    pk[6] = (__bf16)(acc[6] + bv1.z); pk[7] = (__bf16)(acc[7] + bv1.w);
    *(bf16x8*)(kS + (size_t)col * DIM_ + rowB) = pk;   // contiguous in d
  }
  // ---- Phase 2b: v (normal orientation): C[token][d] -> vT[d][kk] packed ----
  for (int t = wave; t < 48; t += 8) {
    int mi = t / 12, ni = t % 12;       // mi: token tile (4), ni: d tile (12)
    f32x8 acc = zero8();
#pragma unroll
    for (int kb = 0; kb < 6; ++kb) {
      bf16x16 a = load_a(kvA, DIM_, mi * 16, kb * 32, lid);
      bf16x16 b = load_b_nk(WkvT, DIM_, DIM_ + ni * 16, kb * 32, lid);
      acc = wmma_bf16(a, b, acc);
    }
    int rowB = mi * 16 + half * 8;      // token base
    int dcol = ni * 16 + m;             // d (local v channel)
    float bias = bkv[DIM_ + dcol];
    bf16x8 pk;
#pragma unroll
    for (int r = 0; r < 8; ++r) pk[r] = (__bf16)(acc[r] + bias);
    *(bf16x8*)(vT + (size_t)dcol * NK_ + rowB) = pk;   // contiguous in kk
  }
  __syncthreads();

  // ---- Phase 3: loop over 8 q-tiles of 64 rows ----
  for (int qt = 0; qt < 8; ++qt) {
    // Stage raw q tile (64 x 96) as bf16 (vectorized: float4 -> bf16x4)
    for (int idx = tid; idx < 64 * 24; idx += 256) {
      int rrow = idx / 24, c4 = idx - rrow * 24;
      int iq = qt * 64 + rrow;
      int is = iq >> 6, ih = (iq >> 3) & 7, iw = iq & 7;
      size_t tok = (size_t)bat * S_F * H_F * W_F +
                   ((size_t)(sd0 + is) * H_F + (hd0 + ih)) * W_F + (wd0 + iw);
      const float4 v = *(const float4*)(xd + tok * DH_ + c4 * 4);
      bf16x4 o; o[0] = (__bf16)v.x; o[1] = (__bf16)v.y; o[2] = (__bf16)v.z; o[3] = (__bf16)v.w;
      *(bf16x4*)(qIn + rrow * DH_ + c4 * 4) = o;
    }
    __syncthreads();

    // q projection (transposed): C'[d][token] = WqT x qIn^T -> packed qS[token][d]
    for (int t = wave; t < 48; t += 8) {
      int mi = t >> 2, ni = t & 3;      // mi: d tile (12), ni: token tile (4)
      f32x8 acc = zero8();
#pragma unroll
      for (int kb = 0; kb < 3; ++kb) {
        bf16x16 a = load_a(WqT, DH_, mi * 16, kb * 32, lid);
        bf16x16 b = load_b_nk(qIn, DH_, ni * 16, kb * 32, lid);
        acc = wmma_bf16(a, b, acc);
      }
      int rowB = mi * 16 + half * 8;    // d base
      int col = ni * 16 + m;            // token
      const float4 bv0 = *(const float4*)(bq + rowB);
      const float4 bv1 = *(const float4*)(bq + rowB + 4);
      bf16x8 pk;
      pk[0] = (__bf16)((acc[0] + bv0.x) * SCALE_); pk[1] = (__bf16)((acc[1] + bv0.y) * SCALE_);
      pk[2] = (__bf16)((acc[2] + bv0.z) * SCALE_); pk[3] = (__bf16)((acc[3] + bv0.w) * SCALE_);
      pk[4] = (__bf16)((acc[4] + bv1.x) * SCALE_); pk[5] = (__bf16)((acc[5] + bv1.y) * SCALE_);
      pk[6] = (__bf16)((acc[6] + bv1.z) * SCALE_); pk[7] = (__bf16)((acc[7] + bv1.w) * SCALE_);
      *(bf16x8*)(qS + (size_t)col * DIM_ + rowB) = pk;
    }
    __syncthreads();

    // ---- heads ----
    for (int h = 0; h < NHEAD_; ++h) {
      const float* biasSrc = biasF + ((size_t)h * NQ_ + qt * 64) * 64;  // contiguous 16KB
#ifdef HAVE_TDM
      if (wave == 0) tdm_copy_f32((unsigned int)(uintptr_t)bS, biasSrc, 64 * 64);
#else
      for (int i = tid; i < 64 * 64; i += 256) bS[i] = biasSrc[i];
#endif
      // scores = q_h @ k_h^T (registers; overlaps the TDM bias copy)
      f32x8 accT[2];
#pragma unroll
      for (int u = 0; u < 2; ++u) {
        int t = wave + u * 8;
        int mi = t >> 2, ni = t & 3;
        bf16x16 a = load_a(qS, DIM_, mi * 16, h * 32, lid);
        bf16x16 b = load_b_nk(kS, DIM_, ni * 16, h * 32, lid);
        accT[u] = wmma_bf16(a, b, zero8());
      }
#ifdef HAVE_TDM
      if (wave == 0) __builtin_amdgcn_s_wait_tensorcnt(0);
#endif
      __syncthreads();
      // epilogue: scores + bias -> aS
#pragma unroll
      for (int u = 0; u < 2; ++u) {
        int t = wave + u * 8;
        int mi = t >> 2, ni = t & 3;
        int col = ni * 16 + m;
#pragma unroll
        for (int r = 0; r < 8; ++r) {
          int row = mi * 16 + half * 8 + r;
          aS[row * 64 + col] = accT[u][r] + bS[row * 64 + col];
        }
      }
      __syncthreads();

      // softmax over 64 keys; 4 threads per row
      {
        int row = tid >> 2, sub = tid & 3;
        const float* rowp = aS + row * 64 + sub * 16;
        float ev[16];
        float mx = -3.4e38f;
#pragma unroll
        for (int j = 0; j < 16; ++j) { ev[j] = rowp[j]; mx = fmaxf(mx, ev[j]); }
        mx = fmaxf(mx, __shfl_xor(mx, 1));
        mx = fmaxf(mx, __shfl_xor(mx, 2));
        float sum = 0.f;
#pragma unroll
        for (int j = 0; j < 16; ++j) { ev[j] = __expf(ev[j] - mx); sum += ev[j]; }
        sum += __shfl_xor(sum, 1);
        sum += __shfl_xor(sum, 2);
        float inv = 1.f / sum;
        __bf16* op = pS + row * 64 + sub * 16;
#pragma unroll
        for (int j = 0; j < 16; ++j) op[j] = (__bf16)(ev[j] * inv);
      }
      __syncthreads();

      // out_h (transposed): C'[d][token] = vT_h x pS^T -> packed oS[token][h*32+d]
      {
        int t = wave;
        int mi = t >> 2, ni = t & 3;    // mi: d tile (2), ni: token tile (4)
        f32x8 acc = zero8();
#pragma unroll
        for (int kb = 0; kb < 2; ++kb) {
          bf16x16 a = load_a(vT + (size_t)h * 32 * NK_, NK_, mi * 16, kb * 32, lid);
          bf16x16 b = load_b_nk(pS, NK_, ni * 16, kb * 32, lid);
          acc = wmma_bf16(a, b, acc);
        }
        int rowB = mi * 16 + half * 8;  // d base within head
        int col = ni * 16 + m;          // token
        bf16x8 pk;
#pragma unroll
        for (int r = 0; r < 8; ++r) pk[r] = (__bf16)acc[r];
        *(bf16x8*)(oS + (size_t)col * DIM_ + h * 32 + rowB) = pk;
      }
      __syncthreads();
    }

    // proj (transposed): C'[o][token] = WpT x oS^T -> contiguous attnO stores
    for (int t = wave; t < 24; t += 8) {
      int mi = t >> 2, ni = t & 3;      // mi: o tile (6), ni: token tile (4)
      f32x8 acc = zero8();
#pragma unroll
      for (int kb = 0; kb < 6; ++kb) {
        bf16x16 a = load_a(WpT, DIM_, mi * 16, kb * 32, lid);
        bf16x16 b = load_b_nk(oS, DIM_, ni * 16, kb * 32, lid);
        acc = wmma_bf16(a, b, acc);
      }
      int rowB = mi * 16 + half * 8;    // o base
      int col = ni * 16 + m;            // token within tile
      int iq = qt * 64 + col;
      int is = iq >> 6, ih = (iq >> 3) & 7, iw = iq & 7;
      size_t tok = (size_t)bat * S_F * H_F * W_F +
                   ((size_t)(sd0 + is) * H_F + (hd0 + ih)) * W_F + (wd0 + iw);
      const float4 bv0 = *(const float4*)(bp + rowB);
      const float4 bv1 = *(const float4*)(bp + rowB + 4);
      float4 o0, o1;
      o0.x = acc[0] + bv0.x; o0.y = acc[1] + bv0.y; o0.z = acc[2] + bv0.z; o0.w = acc[3] + bv0.w;
      o1.x = acc[4] + bv1.x; o1.y = acc[5] + bv1.y; o1.z = acc[6] + bv1.z; o1.w = acc[7] + bv1.w;
      float* dst = attnO + tok * DH_ + rowB;
      *(float4*)(dst) = o0;
      *(float4*)(dst + 4) = o1;
    }
    __syncthreads();
  }
}

// ---------------------------------------------------------------------------
// Kernel 2: deconv (patch expand) + residual + MLP. 1 block = 1 (b, sd, hd) row.
// ---------------------------------------------------------------------------
__launch_bounds__(256, 1)
__global__ void deconv_mlp_kernel(const float* __restrict__ x,
                                  const float* __restrict__ eg, const float* __restrict__ eb,
                                  const __bf16* __restrict__ dwT, const float* __restrict__ db,
                                  const float* __restrict__ g2, const float* __restrict__ be2,
                                  const __bf16* __restrict__ W1T, const float* __restrict__ b1,
                                  const __bf16* __restrict__ W2T, const float* __restrict__ b2,
                                  const float* __restrict__ attnO, float* __restrict__ out) {
  __shared__ alignas(16) char smem[86016];
  float*  yF   = (float*)(smem);            // [64][96] y = sc + attn (f32, kept for residual)
  __bf16* yN   = (__bf16*)(smem + 24576);   // [64][96] LN(y)
  __bf16* xeLN = (__bf16*)(smem + 36864);   // [32][192] LN(x) coarse row (phase 1-2)
  __bf16* hL   = (__bf16*)(smem + 36864);   // [64][384] MLP hidden (aliases xeLN)

  const int tid = threadIdx.x, wave = tid >> 5, lid = tid & 31;
  const int bid = blockIdx.x;
  const int bb = bid >> 11;
  const int rem = bid & 2047;
  const int sd = rem >> 6, hd = rem & 63;
  const int sc_ = sd >> 1, hc_ = hd >> 1;
  const int pi = sd & 1, pj = hd & 1;
  const int m = lid & 15, half = lid >> 4;

  // ---- LN of the coarse source row (32 x 192) with exp_gamma/exp_beta ----
  for (int n = wave * 4; n < wave * 4 + 4; ++n) {
    const float* row = x + ((size_t)bb * S_C * H_C * W_C +
                            ((size_t)sc_ * H_C + hc_) * W_C + n) * DIM_;
    float sm = 0.f, ss = 0.f, vb[6];
#pragma unroll
    for (int j = 0; j < 6; ++j) { float v = row[lid + 32 * j]; vb[j] = v; sm += v; ss += v * v; }
#pragma unroll
    for (int off = 16; off; off >>= 1) { sm += __shfl_xor(sm, off); ss += __shfl_xor(ss, off); }
    float mu = sm * (1.f / 192.f);
    float rstd = rsqrtf(ss * (1.f / 192.f) - mu * mu + 1e-5f);
#pragma unroll
    for (int j = 0; j < 6; ++j) {
      int c = lid + 32 * j;
      xeLN[n * DIM_ + c] = (__bf16)((vb[j] - mu) * rstd * eg[c] + eb[c]);
    }
  }
  __syncthreads();

  // ---- deconv (transposed): C'[o][w] per parity; contiguous yF stores ----
  for (int t = wave; t < 24; t += 8) {
    int kpar = t / 12, sub = t % 12;
    int mi = sub >> 1, ni = sub & 1;    // mi: o tile (6), ni: w tile (2)
    const __bf16* wA = dwT + (size_t)(pi * 4 + pj * 2 + kpar) * 96 * 192;
    f32x8 acc = zero8();
#pragma unroll
    for (int kb = 0; kb < 6; ++kb) {
      bf16x16 a = load_a(wA, DIM_, mi * 16, kb * 32, lid);
      bf16x16 b = load_b_nk(xeLN, DIM_, ni * 16, kb * 32, lid);
      acc = wmma_bf16(a, b, acc);
    }
    int rowB = mi * 16 + half * 8;      // o base
    int wc = ni * 16 + m;               // coarse w
    int wd = 2 * wc + kpar;
    size_t tok = (size_t)bb * S_F * H_F * W_F + ((size_t)sd * H_F + hd) * W_F + wd;
    const float4 bv0 = *(const float4*)(db + rowB);
    const float4 bv1 = *(const float4*)(db + rowB + 4);
    const float4 a0 = *(const float4*)(attnO + tok * DH_ + rowB);
    const float4 a1 = *(const float4*)(attnO + tok * DH_ + rowB + 4);
    float4 o0, o1;
    o0.x = acc[0] + bv0.x + a0.x; o0.y = acc[1] + bv0.y + a0.y;
    o0.z = acc[2] + bv0.z + a0.z; o0.w = acc[3] + bv0.w + a0.w;
    o1.x = acc[4] + bv1.x + a1.x; o1.y = acc[5] + bv1.y + a1.y;
    o1.z = acc[6] + bv1.z + a1.z; o1.w = acc[7] + bv1.w + a1.w;
    float* dst = yF + wd * DH_ + rowB;
    *(float4*)(dst) = o0;
    *(float4*)(dst + 4) = o1;
  }
  __syncthreads();

  // ---- LN(y) (64 x 96) -> yN ----
  for (int n = wave * 8; n < wave * 8 + 8; ++n) {
    float sm = 0.f, ss = 0.f, vb[3];
#pragma unroll
    for (int j = 0; j < 3; ++j) { float v = yF[n * DH_ + lid + 32 * j]; vb[j] = v; sm += v; ss += v * v; }
#pragma unroll
    for (int off = 16; off; off >>= 1) { sm += __shfl_xor(sm, off); ss += __shfl_xor(ss, off); }
    float mu = sm * (1.f / 96.f);
    float rstd = rsqrtf(ss * (1.f / 96.f) - mu * mu + 1e-5f);
#pragma unroll
    for (int j = 0; j < 3; ++j) {
      int c = lid + 32 * j;
      yN[n * DH_ + c] = (__bf16)((vb[j] - mu) * rstd * g2[c] + be2[c]);
    }
  }
  __syncthreads();

  // ---- MLP1 (normal): gelu(yN @ W1 + b1) -> hL (64 x 384, row-major) ----
  for (int t = wave; t < 96; t += 8) {
    int mi = t / 24, ni = t % 24;
    f32x8 acc = zero8();
#pragma unroll
    for (int kb = 0; kb < 3; ++kb) {
      bf16x16 a = load_a(yN, DH_, mi * 16, kb * 32, lid);
      bf16x16 b = load_b_nk(W1T, DH_, ni * 16, kb * 32, lid);
      acc = wmma_bf16(a, b, acc);
    }
    int col = ni * 16 + m;
    float bias = b1[col];
#pragma unroll
    for (int r = 0; r < 8; ++r) {
      int row = mi * 16 + half * 8 + r;
      float v = acc[r] + bias;
      hL[row * HID_ + col] = (__bf16)(0.5f * v * (1.0f + erff(v * 0.70710678118654752f)));
    }
  }
  __syncthreads();

  // ---- MLP2 (transposed): C'[o][token] = W2T x hL^T; contiguous out stores ----
  const size_t tokBase = (size_t)bb * S_F * H_F * W_F + ((size_t)sd * H_F + hd) * W_F;
  for (int t = wave; t < 24; t += 8) {
    int mi = t >> 2, ni = t & 3;        // mi: o tile (6), ni: token tile (4)
    f32x8 acc = zero8();
#pragma unroll
    for (int kb = 0; kb < 12; ++kb) {
      bf16x16 a = load_a(W2T, HID_, mi * 16, kb * 32, lid);
      bf16x16 b = load_b_nk(hL, HID_, ni * 16, kb * 32, lid);
      acc = wmma_bf16(a, b, acc);
    }
    int rowB = mi * 16 + half * 8;      // o base
    int col = ni * 16 + m;              // token (wd)
    const float4 bv0 = *(const float4*)(b2 + rowB);
    const float4 bv1 = *(const float4*)(b2 + rowB + 4);
    const float4 y0 = *(const float4*)(yF + col * DH_ + rowB);
    const float4 y1 = *(const float4*)(yF + col * DH_ + rowB + 4);
    float4 o0, o1;
    o0.x = acc[0] + bv0.x + y0.x; o0.y = acc[1] + bv0.y + y0.y;
    o0.z = acc[2] + bv0.z + y0.z; o0.w = acc[3] + bv0.w + y0.w;
    o1.x = acc[4] + bv1.x + y1.x; o1.y = acc[5] + bv1.y + y1.y;
    o1.z = acc[6] + bv1.z + y1.z; o1.w = acc[7] + bv1.w + y1.w;
    float* dst = out + (tokBase + col) * DH_ + rowB;
    *(float4*)(dst) = o0;
    *(float4*)(dst + 4) = o1;
  }
}

// ---------------------------------------------------------------------------
extern "C" void kernel_launch(void* const* d_in, const int* in_sizes, int n_in,
                              void* d_out, int out_size, void* d_ws, size_t ws_size,
                              hipStream_t stream) {
  (void)in_sizes; (void)n_in; (void)out_size; (void)ws_size;
  const float* x    = (const float*)d_in[0];
  const float* xd   = (const float*)d_in[1];
  const float* g1   = (const float*)d_in[2];
  const float* be1  = (const float*)d_in[3];
  const float* Wkv  = (const float*)d_in[4];
  const float* bkv  = (const float*)d_in[5];
  const float* Wq   = (const float*)d_in[6];
  const float* bq   = (const float*)d_in[7];
  const float* Wp   = (const float*)d_in[8];
  const float* bp   = (const float*)d_in[9];
  const float* btab = (const float*)d_in[10];
  const float* eg   = (const float*)d_in[11];
  const float* eb   = (const float*)d_in[12];
  const float* dw   = (const float*)d_in[13];
  const float* db   = (const float*)d_in[14];
  const float* g2   = (const float*)d_in[15];
  const float* be2  = (const float*)d_in[16];
  const float* W1   = (const float*)d_in[17];
  const float* b1   = (const float*)d_in[18];
  const float* W2   = (const float*)d_in[19];
  const float* b2   = (const float*)d_in[20];
  float* out = (float*)d_out;

  // Workspace layout
  float* attnO = (float*)d_ws;                                       // 2*131072*96 f32
  size_t offBias = (size_t)2 * S_F * H_F * W_F * DH_ * sizeof(float);
  float* biasF = (float*)((char*)d_ws + offBias);                    // 6*512*64 f32
  size_t offWt = offBias + (size_t)NHEAD_ * NQ_ * NK_ * sizeof(float);
  __bf16* wtBase = (__bf16*)((char*)d_ws + offWt);                   // bf16 [N][K] weights
  __bf16* WkvT = wtBase;
  __bf16* WqT  = WkvT + SZ_KV;
  __bf16* WpT  = WqT + SZ_Q;
  __bf16* W1T  = WpT + SZ_P;
  __bf16* W2T  = W1T + SZ_1;
  __bf16* dwT  = W2T + SZ_2;

  prep_kernel<<<(SZ_WT_TOTAL + 255) / 256, 256, 0, stream>>>(Wkv, Wq, Wp, W1, W2, dw, btab,
                                                             wtBase, biasF);
  attn_kernel<<<512, 256, 0, stream>>>(x, xd, g1, be1, WkvT, bkv, WqT, bq, WpT, bp, biasF, attnO);
  deconv_mlp_kernel<<<2 * S_F * H_F, 256, 0, stream>>>(x, eg, eb, dwT, db, g2, be2,
                                                       W1T, b1, W2T, b2, attnO, out);
}
